// SparseGCN_36507222016463
// MI455X (gfx1250) — matrix operand order repaired and checked by reading, not compiled, not run on hardware
//
#include <hip/hip_runtime.h>

// SparseGCN for MI455X (gfx1250, wave32).
// Pipeline: [prep W->bf16^T] -> zero -> spmm-scatter -> WMMA dense+ReLU (x2) -> 128->1 proj.
// Dense layers use v_wmma_f32_16x16x32_bf16 (bf16 A/B, f32 accumulate).

typedef __attribute__((ext_vector_type(16))) __bf16 v16bf;
typedef __attribute__((ext_vector_type(8)))  __bf16 bf16x8;
typedef __attribute__((ext_vector_type(8)))  float  v8f;
typedef __attribute__((ext_vector_type(4)))  float  f32x4;

// ---------------- zero fill ----------------
__global__ void __launch_bounds__(256) zero_f32(float* __restrict__ p, long n) {
  long i = ((long)blockIdx.x * blockDim.x + threadIdx.x) * 4;
  long stride = (long)gridDim.x * blockDim.x * 4;
  for (long j = i; j < n; j += stride) {
    f32x4 z = {0.f, 0.f, 0.f, 0.f};
    *(f32x4*)(p + j) = z;
  }
}

// ---------------- W (f32 [k][n]) -> Wt (bf16 [n][k]) for both layers ----------------
__global__ void __launch_bounds__(256) prep_w(const float* __restrict__ W1,
                                              const float* __restrict__ W2,
                                              __bf16* __restrict__ Wt1,
                                              __bf16* __restrict__ Wt2) {
  int idx = blockIdx.x * 256 + threadIdx.x;        // 0 .. 2*16384-1
  const float* W = (idx < 16384) ? W1 : W2;
  __bf16*     Wt = (idx < 16384) ? Wt1 : Wt2;
  int i = idx & 16383;
  int k = i >> 7;
  int n = i & 127;
  Wt[n * 128 + k] = (__bf16)W[i];
}

// ---------------- COO SpMM: agg[dst] += val * h[src] ----------------
// One wave per edge; lane handles 4 features -> fully coalesced 512B row gather,
// native f32 atomics (global_atomic_add_f32) into the destination row.
__global__ void __launch_bounds__(256) spmm_scatter(const int*   __restrict__ esrc,
                                                    const int*   __restrict__ edst,
                                                    const float* __restrict__ ev,
                                                    const float* __restrict__ h,
                                                    float*       __restrict__ agg,
                                                    int E) {
  int e = blockIdx.x * 8 + (int)(threadIdx.x >> 5);
  if (e >= E) return;
  int lane = threadIdx.x & 31;
  int s = esrc[e];
  int d = edst[e];
  float v = ev[e];
  f32x4 hv = *(const f32x4*)(h + (size_t)s * 128 + lane * 4);
  float* p = agg + (size_t)d * 128 + lane * 4;
  unsafeAtomicAdd(p + 0, v * hv[0]);
  unsafeAtomicAdd(p + 1, v * hv[1]);
  unsafeAtomicAdd(p + 2, v * hv[2]);
  unsafeAtomicAdd(p + 3, v * hv[3]);
}

// ---------------- dense: out = relu(agg @ W + b), 128x128 weight, WMMA bf16 ----------------
// Block: 256 thr = 8 waves; block owns one 16-row M-tile, wave owns one 16-col N-tile.
__global__ void __launch_bounds__(256) dense_relu_wmma(const float*  __restrict__ agg,
                                                       const __bf16* __restrict__ Wt,   // [n][k] bf16
                                                       const float*  __restrict__ bias,
                                                       float*        __restrict__ out,
                                                       int nrows) {
  __shared__ __align__(16) __bf16 ldsA[16 * 128];
  const int m0 = blockIdx.x * 16;
  if (m0 >= nrows) return;
  const int t = threadIdx.x;

  // Stage the 16x128 f32 A-tile into LDS as bf16 (8 elements / thread).
  {
    const int row = t >> 4;
    const int col = (t & 15) * 8;
    const float* src = agg + (size_t)(m0 + row) * 128 + col;
    f32x4 a0 = *(const f32x4*)src;
    f32x4 a1 = *(const f32x4*)(src + 4);
    bf16x8 v;
    v[0] = (__bf16)a0[0]; v[1] = (__bf16)a0[1]; v[2] = (__bf16)a0[2]; v[3] = (__bf16)a0[3];
    v[4] = (__bf16)a1[0]; v[5] = (__bf16)a1[1]; v[6] = (__bf16)a1[2]; v[7] = (__bf16)a1[3];
    *(bf16x8*)(ldsA + row * 128 + col) = v;
  }
  __syncthreads();

  const int wave = t >> 5;          // 0..7  -> N-tile
  const int lane = t & 31;
  const int half = lane >> 4;       // 0/1
  const int l16  = lane & 15;
  const int n0   = wave * 16;

  // A frag (16x32 bf16): lane = row l16; K groups {kbase..kbase+7} and {+16..+23}, kbase=8*half.
  const __bf16* arow = ldsA + l16 * 128 + half * 8;
  // B frag (32x16 bf16): lane = col n0+l16; K groups {kbase..kbase+7} and {+16..+23}, kbase=8*half.
  const __bf16* wrow = Wt + (size_t)(n0 + l16) * 128 + half * 8;

  v8f acc = {};
#pragma unroll
  for (int ks = 0; ks < 4; ++ks) {
    bf16x8 alo = *(const bf16x8*)(arow + ks * 32);
    bf16x8 ahi = *(const bf16x8*)(arow + ks * 32 + 16);
    bf16x8 blo = *(const bf16x8*)(wrow + ks * 32);
    bf16x8 bhi = *(const bf16x8*)(wrow + ks * 32 + 16);
    v16bf A = __builtin_shufflevector(alo, ahi, 0,1,2,3,4,5,6,7,8,9,10,11,12,13,14,15);
    v16bf B = __builtin_shufflevector(blo, bhi, 0,1,2,3,4,5,6,7,8,9,10,11,12,13,14,15);
    acc = __builtin_amdgcn_wmma_f32_16x16x32_bf16(false, A, false, B,
                                                  (short)0, acc, false, false);
  }

  // C/D layout: VGPR r -> row m0 + 8*half + r, col n0 + l16. Bias + ReLU epilogue.
  const int   n  = n0 + l16;
  const float bn = bias[n];
#pragma unroll
  for (int r = 0; r < 8; ++r) {
    const int m = m0 + 8 * half + r;
    float val = acc[r] + bn;
    out[(size_t)m * 128 + n] = val > 0.f ? val : 0.f;
  }
}

// ---------------- output projection: out[i] = h[i,:] . Wout + bout ----------------
__global__ void __launch_bounds__(256) out_proj(const float* __restrict__ h,
                                                const float* __restrict__ Wout,
                                                const float* __restrict__ bout,
                                                float*       __restrict__ out,
                                                int nrows) {
  int row = blockIdx.x * 8 + (int)(threadIdx.x >> 5);
  if (row >= nrows) return;
  int lane = threadIdx.x & 31;
  f32x4 hv = *(const f32x4*)(h + (size_t)row * 128 + lane * 4);
  f32x4 wv = *(const f32x4*)(Wout + lane * 4);
  float p = hv[0] * wv[0] + hv[1] * wv[1] + hv[2] * wv[2] + hv[3] * wv[3];
#pragma unroll
  for (int off = 16; off; off >>= 1) p += __shfl_xor(p, off, 32);
  if (lane == 0) out[row] = p + bout[0];
}

extern "C" void kernel_launch(void* const* d_in, const int* in_sizes, int n_in,
                              void* d_out, int out_size, void* d_ws, size_t ws_size,
                              hipStream_t stream) {
  const float* x    = (const float*)d_in[0];
  const int*   esrc = (const int*)  d_in[1];
  const int*   edst = (const int*)  d_in[2];
  const float* ev   = (const float*)d_in[3];
  const float* W1   = (const float*)d_in[4];
  const float* b1   = (const float*)d_in[5];
  const float* W2   = (const float*)d_in[6];
  const float* b2   = (const float*)d_in[7];
  const float* Wout = (const float*)d_in[8];
  const float* bout = (const float*)d_in[9];
  float* out = (float*)d_out;

  const int N = in_sizes[0] / 128;
  const int E = in_sizes[1];

  char*   ws  = (char*)d_ws;
  float*  agg = (float*)ws;                                        // N*128 f32
  float*  h   = (float*)(ws + (size_t)N * 128 * sizeof(float));    // N*128 f32
  __bf16* Wt1 = (__bf16*)(ws + 2 * (size_t)N * 128 * sizeof(float));
  __bf16* Wt2 = Wt1 + 128 * 128;

  const int zeroBlocks  = 4096;
  const int spmmBlocks  = (E + 7) / 8;
  const int denseBlocks = (N + 15) / 16;
  const int outBlocks   = (N + 7) / 8;

  prep_w<<<128, 256, 0, stream>>>(W1, W2, Wt1, Wt2);

  // Layer 1
  zero_f32<<<zeroBlocks, 256, 0, stream>>>(agg, (long)N * 128);
  spmm_scatter<<<spmmBlocks, 256, 0, stream>>>(esrc, edst, ev, x, agg, E);
  dense_relu_wmma<<<denseBlocks, 256, 0, stream>>>(agg, Wt1, b1, h, N);

  // Layer 2
  zero_f32<<<zeroBlocks, 256, 0, stream>>>(agg, (long)N * 128);
  spmm_scatter<<<spmmBlocks, 256, 0, stream>>>(esrc, edst, ev, h, agg, E);
  dense_relu_wmma<<<denseBlocks, 256, 0, stream>>>(agg, Wt2, b2, h, N);

  // Output projection
  out_proj<<<outBlocks, 256, 0, stream>>>(h, Wout, bout, out, N);
}
